// MatchingProbModule_36189394436349
// MI455X (gfx1250) — matrix operand order repaired and checked by reading, not compiled, not run on hardware
//
#include <hip/hip_runtime.h>

// ---------------------------------------------------------------------------
// MatchingProb on MI455X (gfx1250):
//   1) prep: fp32 -> bf16 row copies + row squared-norms
//   2) WMMA bf16 GEMM (v_wmma_f32_16x16x32_bf16), K-tiles streamed into LDS by
//      the Tensor Data Mover (double-buffered, s_wait_tensorcnt pipelined),
//      fused dist/sigmoid epilogue
//   3) per-row 16K bitonic argsort fully in LDS (128KB/WG; CDNA5 320KB WGP LDS)
// ---------------------------------------------------------------------------

typedef __attribute__((ext_vector_type(8)))  __bf16 v8bf;
typedef __attribute__((ext_vector_type(16))) __bf16 v16bf;
typedef __attribute__((ext_vector_type(8)))  float  v8f;
typedef __attribute__((ext_vector_type(4)))  unsigned int u32x4;
typedef __attribute__((ext_vector_type(8)))  int i32x8;
typedef __attribute__((ext_vector_type(4)))  int i32x4;

#define BR 128      // block tile rows (queries)
#define BC 128      // block tile cols (gallery)
#define BK 32       // K per WMMA step (bf16)
#define LDSK 40     // LDS row stride in elements (80B: 16B-aligned, bank-spread)

#if defined(__AMDGCN__) && __has_builtin(__builtin_amdgcn_tensor_load_to_lds) && \
    __has_builtin(__builtin_amdgcn_s_wait_tensorcnt)
#define ATHENA_TDM 1
#else
#define ATHENA_TDM 0
#endif

__device__ __forceinline__ unsigned short f2bf(float f) {
  unsigned u = __float_as_uint(f);
  u += 0x7FFFu + ((u >> 16) & 1u);           // round-to-nearest-even
  return (unsigned short)(u >> 16);
}

// One block per row of X[.,D]; D == 1024, blockDim == 256 (one float4/thread).
__global__ __launch_bounds__(256)
void prep_rows_kernel(const float* __restrict__ X, unsigned short* __restrict__ Xb,
                      float* __restrict__ x2, int D) {
  const int row = blockIdx.x;
  const int tid = threadIdx.x;
  const float4 v = ((const float4*)(X + (size_t)row * D))[tid];
  float s = v.x * v.x + v.y * v.y + v.z * v.z + v.w * v.w;
  uint2 p;
  p.x = (unsigned)f2bf(v.x) | ((unsigned)f2bf(v.y) << 16);
  p.y = (unsigned)f2bf(v.z) | ((unsigned)f2bf(v.w) << 16);
  ((uint2*)(Xb + (size_t)row * D))[tid] = p;

  s += __shfl_down(s, 16, 32);
  s += __shfl_down(s, 8, 32);
  s += __shfl_down(s, 4, 32);
  s += __shfl_down(s, 2, 32);
  s += __shfl_down(s, 1, 32);
  __shared__ float red[8];
  if ((tid & 31) == 0) red[tid >> 5] = s;
  __syncthreads();
  if (tid < 32) {
    float t = (tid < 8) ? red[tid] : 0.0f;
    t += __shfl_down(t, 4, 32);
    t += __shfl_down(t, 2, 32);
    t += __shfl_down(t, 1, 32);
    if (tid == 0) x2[row] = t;
  }
}

#if ATHENA_TDM
// Issue one TDM load of a [nrows x 32] bf16 tile (row stride D elements in
// memory) into LDS at lds_addr. D# padding (+16B per 64B row) lands the tile
// with an 80B row stride == LDSK, matching the WMMA fragment layout.
__device__ __forceinline__ void tdm_load_tile(const unsigned short* base, int row0,
                                              int k0, unsigned lds_addr, int D,
                                              int nrows) {
  unsigned long long ga = (unsigned long long)(size_t)base +
      (((unsigned long long)(unsigned)row0 * (unsigned)D + (unsigned)k0) << 1);
  u32x4 g0;
  g0[0] = 1u;                                        // count=1 (valid), no gather
  g0[1] = lds_addr;                                  // LDS byte address
  g0[2] = (unsigned)ga;                              // global_addr[31:0]
  g0[3] = (unsigned)((ga >> 32) & 0x01FFFFFFull) |   // global_addr[56:32]
          (2u << 30);                                // type=2 ("image")
  i32x8 g1;
  g1[0] = (int)((1u << 16) |                         // data_size = 2B
                (1u << 20) |                         // pad_enable
                (3u << 22) |                         // pad_interval: 16 DWORDs (64B)
                (3u << 25));                         // pad_amount: 4 DWORDs (16B)
  g1[1] = (int)(((unsigned)D & 0xFFFFu) << 16);      // tensor_dim0[15:0]
  g1[2] = (int)(((unsigned)D >> 16) |                // tensor_dim0[31:16]
                ((unsigned)nrows << 16));            // tensor_dim1[15:0]
  g1[3] = (int)(32u << 16);                          // tile_dim0 = 32
  g1[4] = nrows;                                     // tile_dim1
  g1[5] = D;                                         // tensor_dim0_stride[31:0]
  g1[6] = 0;
  g1[7] = 0;
  i32x4 z4 = {0, 0, 0, 0};
#if __clang_major__ >= 23
  i32x8 z8 = {0, 0, 0, 0, 0, 0, 0, 0};
  __builtin_amdgcn_tensor_load_to_lds(g0, g1, z4, z4, z8, 0);
#else
  __builtin_amdgcn_tensor_load_to_lds(g0, g1, z4, z4, 0);
#endif
}
#endif  // ATHENA_TDM

// 128x128 block tile, 256 threads = 8 waves (wave32), each wave owns 64x32.
// Per K-step: 4 A-frags x 2 B-frags -> 8 v_wmma_f32_16x16x32_bf16.
__global__ __launch_bounds__(256)
void wmma_dist_kernel(const unsigned short* __restrict__ Qb,
                      const unsigned short* __restrict__ Gb,
                      const float* __restrict__ q2, const float* __restrict__ g2,
                      const float* __restrict__ pa, const float* __restrict__ pb,
                      float* __restrict__ outv, int R, int C, int D) {
  __shared__ __bf16 As[2][BR * LDSK];
  __shared__ __bf16 Bs[2][BC * LDSK];

  const int tid   = threadIdx.x;
  const int lane  = tid & 31;
  const int wave  = tid >> 5;       // 0..7
  const int wr    = wave >> 2;      // 0..1 : row half of the block tile
  const int wc    = wave & 3;       // 0..3 : column quarter
  const int lrow  = lane & 15;
  const int lhalf = lane >> 4;

  const int r0 = blockIdx.y * BR;
  const int c0 = blockIdx.x * BC;

  v8f acc[4][2] = {};
  const int kA0 = lhalf ? 8 : 0;    // A-frag K base per lane half
  int p = 0;

#if ATHENA_TDM
  const unsigned ldsA0 = (unsigned)(size_t)&As[0][0];
  const unsigned ldsA1 = (unsigned)(size_t)&As[1][0];
  const unsigned ldsB0 = (unsigned)(size_t)&Bs[0][0];
  const unsigned ldsB1 = (unsigned)(size_t)&Bs[1][0];
  // preload K-tile 0 into buffer 0 (wave0: A, wave1: B), DMA via TDM
  if (wave == 0)      tdm_load_tile(Qb, r0, 0, ldsA0, D, BR);
  else if (wave == 1) tdm_load_tile(Gb, c0, 0, ldsB0, D, BC);
  if (wave < 2) __builtin_amdgcn_s_wait_tensorcnt(0);
  __syncthreads();
#else
  // manual tile loaders: 2 threads per 32-element row, 16 bf16 (2x uint4) each
  const int ldrow = tid >> 1;       // 0..127
  const int ldseg = tid & 1;        // 0,1
  const size_t aGlobalBase = (size_t)(r0 + ldrow) * D + (size_t)ldseg * 16;
  const size_t bGlobalBase = (size_t)(c0 + ldrow) * D + (size_t)ldseg * 16;
  uint4* aDst = (uint4*)&As[0][ldrow * LDSK + ldseg * 16];
  uint4* bDst = (uint4*)&Bs[0][ldrow * LDSK + ldseg * 16];
#endif

  for (int k0 = 0; k0 < D; k0 += BK) {
#if ATHENA_TDM
    // kick DMA of the next K-tile into the idle buffer while we compute
    if (k0 + BK < D) {
      if (wave == 0)      tdm_load_tile(Qb, r0, k0 + BK, p ? ldsA0 : ldsA1, D, BR);
      else if (wave == 1) tdm_load_tile(Gb, c0, k0 + BK, p ? ldsB0 : ldsB1, D, BC);
    }
#else
    const uint4* aSrc = (const uint4*)(Qb + aGlobalBase + k0);
    const uint4* bSrc = (const uint4*)(Gb + bGlobalBase + k0);
    uint4 aq0 = aSrc[0], aq1 = aSrc[1];
    uint4 bq0 = bSrc[0], bq1 = bSrc[1];
    if (k0 + BK < D) {
      __builtin_prefetch(Qb + aGlobalBase + k0 + BK, 0, 1);
      __builtin_prefetch(Gb + bGlobalBase + k0 + BK, 0, 1);
    }
    __syncthreads();                 // previous iteration's LDS reads done
    aDst[0] = aq0; aDst[1] = aq1;
    bDst[0] = bq0; bDst[1] = bq1;
    __syncthreads();
#endif

    const __bf16* Abase = As[p];
    const __bf16* Bbase = Bs[p];

    v16bf afrag[4];
#pragma unroll
    for (int rt = 0; rt < 4; ++rt) {
      const __bf16* ab = &Abase[(wr * 64 + rt * 16 + lrow) * LDSK + kA0];
      v8bf lo = *(const v8bf*)ab;          // K = kA0..kA0+7
      v8bf hi = *(const v8bf*)(ab + 16);   // K = kA0+16..kA0+23
      afrag[rt] = __builtin_shufflevector(lo, hi, 0, 1, 2, 3, 4, 5, 6, 7,
                                          8, 9, 10, 11, 12, 13, 14, 15);
    }
    v16bf bfrag[2];
#pragma unroll
    for (int ct = 0; ct < 2; ++ct) {
      const __bf16* bb = &Bbase[(wc * 32 + ct * 16 + lrow) * LDSK + lhalf * 8];
      v8bf lo = *(const v8bf*)bb;          // K = 8h..8h+7, then 16+8h..16+8h+7
      v8bf hi = *(const v8bf*)(bb + 16);
      bfrag[ct] = __builtin_shufflevector(lo, hi, 0, 1, 2, 3, 4, 5, 6, 7,
                                          8, 9, 10, 11, 12, 13, 14, 15);
    }

#pragma unroll
    for (int rt = 0; rt < 4; ++rt)
#pragma unroll
      for (int ct = 0; ct < 2; ++ct)
        acc[rt][ct] = __builtin_amdgcn_wmma_f32_16x16x32_bf16(
            false, afrag[rt], false, bfrag[ct], (short)0, acc[rt][ct],
            false, false);

#if ATHENA_TDM
    if (k0 + BK < D && wave < 2) __builtin_amdgcn_s_wait_tensorcnt(0);
    __syncthreads();                 // publish next buffer / retire this one
    p ^= 1;
#endif
  }

  // fused epilogue: d2 = q2 + g2 - 2*dot ; neg = -sigmoid(-a*dist + b)
  const float a_s = pa[0];
  const float b_s = pb[0];
#pragma unroll
  for (int rt = 0; rt < 4; ++rt) {
#pragma unroll
    for (int ct = 0; ct < 2; ++ct) {
      const int col = c0 + wc * 32 + ct * 16 + lrow;
      const float gn = g2[col];
#pragma unroll
      for (int v = 0; v < 8; ++v) {
        const int row = r0 + wr * 64 + rt * 16 + v + lhalf * 8;
        float d2 = q2[row] + gn - 2.0f * acc[rt][ct][v];
        d2 = fmaxf(d2, 0.0f);
        const float dist = __builtin_sqrtf(d2);
        const float sim = 1.0f / (1.0f + __expf(a_s * dist - b_s));
        outv[(size_t)row * C + col] = -sim;
      }
    }
  }
}

// One block per output row: stable ascending argsort of C=16K elements,
// entirely in LDS (C * 8B = 128KB dynamic shared; CDNA5 WGP has 320KB).
// Key = order-preserving float bits (hi32) | index (lo32) -> stable ties.
__global__ __launch_bounds__(512)
void bitonic_sort_kernel(float* __restrict__ outv, float* __restrict__ outi, int C) {
  extern __shared__ unsigned long long keys[];
  const int row = blockIdx.x;
  const int tid = threadIdx.x;
  const int nt  = blockDim.x;
  float* vals = outv + (size_t)row * C;

  for (int i = tid; i < C; i += nt) {
    unsigned u = __float_as_uint(vals[i]);
    u = (u >> 31) ? ~u : (u | 0x80000000u);   // order-preserving transform
    keys[i] = ((unsigned long long)u << 32) | (unsigned)i;
  }
  __syncthreads();

  for (int k = 2; k <= C; k <<= 1) {
    for (int j = k >> 1; j > 0; j >>= 1) {
      for (int i = tid; i < C; i += nt) {
        const int ixj = i ^ j;
        if (ixj > i) {
          const unsigned long long x = keys[i];
          const unsigned long long y = keys[ixj];
          const bool up = ((i & k) == 0);
          if (up ? (x > y) : (x < y)) { keys[i] = y; keys[ixj] = x; }
        }
      }
      __syncthreads();
    }
  }

  for (int i = tid; i < C; i += nt) {
    const unsigned long long kk = keys[i];
    unsigned u = (unsigned)(kk >> 32);
    u = (u & 0x80000000u) ? (u ^ 0x80000000u) : ~u;  // inverse transform
    vals[i] = __uint_as_float(u);
    outi[(size_t)row * C + i] = (float)(unsigned)(kk & 0xFFFFFFFFu);
  }
}

extern "C" void kernel_launch(void* const* d_in, const int* in_sizes, int n_in,
                              void* d_out, int out_size, void* d_ws, size_t ws_size,
                              hipStream_t stream) {
  const int D = 1024;
  const int R = in_sizes[0] / D;     // 4096 queries
  const int C = in_sizes[1] / D;     // 16384 gallery

  const float* q  = (const float*)d_in[0];
  const float* g  = (const float*)d_in[1];
  const float* pa = (const float*)d_in[2];
  const float* pb = (const float*)d_in[3];

  float* outv = (float*)d_out;                  // sorted -sims (ascending)
  float* outi = outv + (size_t)R * C;           // argsort indices (as floats)

  // Scratch (bf16 copies + norms, ~40MB) lives in the index half of d_out;
  // it is consumed before the sort phase rewrites that region. Every byte
  // read is written earlier in this same launch sequence (graph-replay safe).
  unsigned short* Qb = (unsigned short*)outi;
  unsigned short* Gb = Qb + (size_t)R * D;
  float* q2 = (float*)(Gb + (size_t)C * D);
  float* g2 = q2 + R;

  prep_rows_kernel<<<R, 256, 0, stream>>>(q, Qb, q2, D);
  prep_rows_kernel<<<C, 256, 0, stream>>>(g, Gb, g2, D);

  dim3 grid(C / BC, R / BR);
  wmma_dist_kernel<<<grid, 256, 0, stream>>>(Qb, Gb, q2, g2, pa, pb, outv, R, C, D);

  const size_t lds_bytes = (size_t)C * sizeof(unsigned long long);  // 128KB
  hipFuncSetAttribute((const void*)bitonic_sort_kernel,
                      hipFuncAttributeMaxDynamicSharedMemorySize, (int)lds_bytes);
  bitonic_sort_kernel<<<R, 512, lds_bytes, stream>>>(outv, outi, C);
}
